// MultiHeadAttention_25202868093129
// MI455X (gfx1250) — compile-verified, matrix operand order checked by reading, not run on hardware
//
#include <hip/hip_runtime.h>
#include <hip/hip_bf16.h>
#include <stdint.h>

// ---------------------------------------------------------------------------
// MHA forward for MI455X (gfx1250, wave32, WMMA).
// fp32 reference approximated with bf16 WMMA (f32 accumulate).
// ~69 GFLOP of matrix math vs ~50MB true HBM traffic @23.3TB/s -> entirely
// matrix-op bound; everything runs on v_wmma_f32_16x16x32_bf16.
// ---------------------------------------------------------------------------

#define HIDDEN   1024
#define HEADS    16
#define HEAD_DIM 64
#define BATCH    2
#define SEQ      2048
#define MROWS    (BATCH * SEQ)   // 4096
#define ATT_SCALE 0.125f         // 1/sqrt(64), folded into Q projection

typedef __attribute__((ext_vector_type(16))) __bf16 v16bf;
typedef __attribute__((ext_vector_type(8)))  float  v8f;

__device__ __forceinline__ __bf16 f2bf(float f) {
  union { float f; uint32_t u; } a; a.f = f;
  uint32_t r = a.u + 0x7fffu + ((a.u >> 16) & 1u);   // round-to-nearest-even
  union { unsigned short s; __bf16 b; } o;
  o.s = (unsigned short)(r >> 16);
  return o.b;
}

__device__ __forceinline__ v8f zero8() {
  v8f z = {0.f, 0.f, 0.f, 0.f, 0.f, 0.f, 0.f, 0.f};
  return z;
}

__device__ __forceinline__ v8f wmma_bf16(v16bf a, v16bf b, v8f c) {
  // (neg_a, A, neg_b, B, c_mod, C, reuse_a, reuse_b)
  return __builtin_amdgcn_wmma_f32_16x16x32_bf16(false, a, false, b,
                                                 (short)0, c, false, false);
}

// Load a 16x32 bf16 A/B fragment from row-major storage (K contiguous).
// WMMA 16-bit operand layout: lane = row%16, lanes 0-15 hold K {0..7,16..23},
// lanes 16-31 hold K {8..15,24..31} -> two contiguous 16-byte loads per lane.
__device__ __forceinline__ v16bf load_frag(const __bf16* __restrict__ base,
                                           int row0, int ld, int k0, int lane) {
  const int r = row0 + (lane & 15);
  const int h = lane >> 4;
  const __bf16* p = base + (size_t)r * ld + (k0 + h * 8);
  v16bf out;
  ((uint4*)&out)[0] = *(const uint4*)p;         // K = k0 + 8h   .. +7
  ((uint4*)&out)[1] = *(const uint4*)(p + 16);  // K = k0+16+8h  .. +7
  return out;
}

// ---------------------------------------------------------------------------
// Pre-pass conversions
// ---------------------------------------------------------------------------
__global__ void convert_x_kernel(const float* __restrict__ x,
                                 __bf16* __restrict__ xb, int n) {
  int i = blockIdx.x * blockDim.x + threadIdx.x;
  if (i < n) xb[i] = f2bf(x[i]);
}

// wt[n*1024 + k] = bf16(w[k*1024 + n])  (store W^T so B-fragments are contiguous)
__global__ void transpose_w_kernel(const float* __restrict__ w,
                                   __bf16* __restrict__ wt) {
  int idx = blockIdx.x * blockDim.x + threadIdx.x;   // over 1024*1024
  int n = idx >> 10;
  int k = idx & 1023;
  wt[idx] = f2bf(w[(size_t)k * HIDDEN + n]);
}

// ---------------------------------------------------------------------------
// GEMM: C[M=4096, N=1024] = (A(bf16,[M,K]) @ Bt(bf16,[N,K])^T + bias) * scale
// MODE 0: write bf16 to [B,H,S,Hd]          (Q, K projections)
// MODE 1: write bf16 to [B,H,Hd,S]          (V projection, transposed)
// MODE 2: write fp32 row-major [M,N]        (output projection -> d_out)
// Block = 256 threads = 8 waves (2x4); wave tile = 64x32 (4x2 WMMA tiles).
// ---------------------------------------------------------------------------
template <int MODE>
__global__ __launch_bounds__(256)
void gemm_bf16_kernel(const __bf16* __restrict__ A,
                      const __bf16* __restrict__ Bt,
                      const float* __restrict__ bias,
                      void* __restrict__ out,
                      float store_scale) {
  const int lane  = threadIdx.x & 31;
  const int wid   = threadIdx.x >> 5;
  const int waveM = wid >> 2;            // 0..1
  const int waveN = wid & 3;             // 0..3
  const int m0 = blockIdx.x * 128 + waveM * 64;
  const int n0 = blockIdx.y * 128 + waveN * 32;

  v8f acc[4][2];
#pragma unroll
  for (int i = 0; i < 4; ++i)
#pragma unroll
    for (int j = 0; j < 2; ++j) acc[i][j] = zero8();

  const int lrow = lane & 15;
  const int lhalf = lane >> 4;

  for (int k0 = 0; k0 < HIDDEN; k0 += 32) {
    // Hint the next K-slice into cache (global_prefetch_b8 on gfx1250).
    if (k0 + 32 < HIDDEN) {
      __builtin_prefetch(A + (size_t)(m0 + lrow) * HIDDEN + (k0 + 32) + lhalf * 8, 0, 1);
      __builtin_prefetch(Bt + (size_t)(n0 + lrow) * HIDDEN + (k0 + 32) + lhalf * 8, 0, 1);
    }
    v16bf af[4], bfr[2];
#pragma unroll
    for (int i = 0; i < 4; ++i)
      af[i] = load_frag(A, m0 + i * 16, HIDDEN, k0, lane);
#pragma unroll
    for (int j = 0; j < 2; ++j)
      bfr[j] = load_frag(Bt, n0 + j * 16, HIDDEN, k0, lane);
#pragma unroll
    for (int i = 0; i < 4; ++i)
#pragma unroll
      for (int j = 0; j < 2; ++j)
        acc[i][j] = wmma_bf16(af[i], bfr[j], acc[i][j]);
  }

  // C/D layout: lane holds column n = n0 + j*16 + (lane&15);
  // VGPR r holds row m = r + 8*(lane>>4).
#pragma unroll
  for (int i = 0; i < 4; ++i) {
#pragma unroll
    for (int j = 0; j < 2; ++j) {
      const int n = n0 + j * 16 + lrow;
      const float bv = bias[n];
#pragma unroll
      for (int r = 0; r < 8; ++r) {
        const int m = m0 + i * 16 + r + 8 * lhalf;
        if (MODE == 2) {
          ((float*)out)[(size_t)m * HIDDEN + n] = acc[i][j][r] + bv;
        } else {
          const float val = (acc[i][j][r] + bv) * store_scale;
          const int b  = m >> 11;      // m / SEQ
          const int s  = m & (SEQ - 1);
          const int hh = n >> 6;       // head
          const int d  = n & 63;
          if (MODE == 0) {
            ((__bf16*)out)[(((size_t)b * HEADS + hh) * SEQ + s) * HEAD_DIM + d] =
                f2bf(val);
          } else {
            ((__bf16*)out)[(((size_t)b * HEADS + hh) * HEAD_DIM + d) * SEQ + s] =
                f2bf(val);
          }
        }
      }
    }
  }
}

// ---------------------------------------------------------------------------
// Flash attention: grid = (SEQ/128, B*H), block = 256 (8 waves).
// Each wave owns 16 query rows; K-tiles of 64 keys (8 score WMMAs + 8 PV
// WMMAs per online-softmax pass to amortize the shuffle reductions).
// Q is pre-scaled by 1/sqrt(Hd) in the projection.
// ---------------------------------------------------------------------------
__global__ __launch_bounds__(256)
void attn_kernel(const __bf16* __restrict__ Q,   // [B,H,S,Hd] (pre-scaled)
                 const __bf16* __restrict__ K,   // [B,H,S,Hd]
                 const __bf16* __restrict__ Vt,  // [B,H,Hd,S]
                 __bf16* __restrict__ ctx) {     // [B,S,H*Hd]
  __shared__ __align__(16) __bf16 pbuf[8][16 * 64];  // per-wave P tile (16KB)

  const int lane = threadIdx.x & 31;
  const int wid  = threadIdx.x >> 5;
  const int bh = blockIdx.y;
  const int b = bh >> 4, h = bh & 15;
  const int q0 = blockIdx.x * 128 + wid * 16;

  const __bf16* qb = Q  + ((size_t)b * HEADS + h) * SEQ * HEAD_DIM;
  const __bf16* kb = K  + ((size_t)b * HEADS + h) * SEQ * HEAD_DIM;
  const __bf16* vb = Vt + ((size_t)b * HEADS + h) * HEAD_DIM * SEQ;

  // Q fragments: 16 rows x 64 (two K=32 fragments), loaded once.
  const v16bf qf0 = load_frag(qb, q0, HEAD_DIM, 0, lane);
  const v16bf qf1 = load_frag(qb, q0, HEAD_DIM, 32, lane);

  v8f o[4];
#pragma unroll
  for (int j = 0; j < 4; ++j) o[j] = zero8();
  float mrow[8], lsum[8];
#pragma unroll
  for (int r = 0; r < 8; ++r) { mrow[r] = -3.0e38f; lsum[r] = 0.0f; }

  const int half = lane >> 4;
  const int col  = lane & 15;

  for (int kt = 0; kt < SEQ; kt += 64) {
    // Scores for 64 keys: four 16x16 tiles, two K=32 WMMAs each.
    v8f s[4];
#pragma unroll
    for (int t = 0; t < 4; ++t) {
      v16bf kf0 = load_frag(kb, kt + t * 16, HEAD_DIM, 0,  lane);
      v16bf kf1 = load_frag(kb, kt + t * 16, HEAD_DIM, 32, lane);
      v8f a = zero8();
      a = wmma_bf16(qf0, kf0, a);
      a = wmma_bf16(qf1, kf1, a);
      s[t] = a;
    }

    // Online softmax (rows live in (r, half); 16 lanes per half share a row).
#pragma unroll
    for (int r = 0; r < 8; ++r) {
      float mx = fmaxf(fmaxf(s[0][r], s[1][r]), fmaxf(s[2][r], s[3][r]));
#pragma unroll
      for (int off = 8; off >= 1; off >>= 1)
        mx = fmaxf(mx, __shfl_xor(mx, off, 32));   // stays within 16-lane half
      const float mnew = fmaxf(mrow[r], mx);
      const float alpha = __expf(mrow[r] - mnew);
      mrow[r] = mnew;
      const float p0 = __expf(s[0][r] - mnew);
      const float p1 = __expf(s[1][r] - mnew);
      const float p2 = __expf(s[2][r] - mnew);
      const float p3 = __expf(s[3][r] - mnew);
      s[0][r] = p0; s[1][r] = p1; s[2][r] = p2; s[3][r] = p3;
      float ps = (p0 + p1) + (p2 + p3);
#pragma unroll
      for (int off = 8; off >= 1; off >>= 1)
        ps += __shfl_xor(ps, off, 32);
      lsum[r] = lsum[r] * alpha + ps;
#pragma unroll
      for (int j = 0; j < 4; ++j) o[j][r] = o[j][r] * alpha;
    }

    // Convert P (16x64, C-layout) -> two A fragments via per-wave LDS bounce.
    __syncthreads();
#pragma unroll
    for (int r = 0; r < 8; ++r) {
      const int m = r + 8 * half;
      pbuf[wid][m * 64 + col]      = f2bf(s[0][r]);
      pbuf[wid][m * 64 + 16 + col] = f2bf(s[1][r]);
      pbuf[wid][m * 64 + 32 + col] = f2bf(s[2][r]);
      pbuf[wid][m * 64 + 48 + col] = f2bf(s[3][r]);
    }
    __syncthreads();
    v16bf pa0, pa1;
    {
      const __bf16* p = &pbuf[wid][(lane & 15) * 64 + half * 8];
      ((uint4*)&pa0)[0] = *(const uint4*)p;          // keys kt+ {8h..8h+7}
      ((uint4*)&pa0)[1] = *(const uint4*)(p + 16);   // keys kt+16+8h..
      ((uint4*)&pa1)[0] = *(const uint4*)(p + 32);   // keys kt+32+8h..
      ((uint4*)&pa1)[1] = *(const uint4*)(p + 48);   // keys kt+48+8h..
    }

    // O += P @ V   (Vt rows = d, keys contiguous -> direct B fragments)
#pragma unroll
    for (int j = 0; j < 4; ++j) {
      v16bf vf0 = load_frag(vb, j * 16, SEQ, kt,      lane);
      v16bf vf1 = load_frag(vb, j * 16, SEQ, kt + 32, lane);
      o[j] = wmma_bf16(pa0, vf0, o[j]);
      o[j] = wmma_bf16(pa1, vf1, o[j]);
    }
  }

  // Normalize (one reciprocal per row) and scatter to ctx [B,S,H*Hd].
#pragma unroll
  for (int r = 0; r < 8; ++r) {
    const int s = q0 + r + 8 * half;
    const float inv = 1.0f / lsum[r];
#pragma unroll
    for (int j = 0; j < 4; ++j) {
      const int d = j * 16 + col;
      ctx[((size_t)b * SEQ + s) * HIDDEN + h * HEAD_DIM + d] =
          f2bf(o[j][r] * inv);
    }
  }
}

// ---------------------------------------------------------------------------
// Host launch
// ---------------------------------------------------------------------------
extern "C" void kernel_launch(void* const* d_in, const int* in_sizes, int n_in,
                              void* d_out, int out_size, void* d_ws, size_t ws_size,
                              hipStream_t stream) {
  const float* x  = (const float*)d_in[0];
  const float* wq = (const float*)d_in[1];
  const float* bq = (const float*)d_in[2];
  const float* wk = (const float*)d_in[3];
  const float* bk = (const float*)d_in[4];
  const float* wv = (const float*)d_in[5];
  const float* bv = (const float*)d_in[6];
  const float* wo = (const float*)d_in[7];
  const float* bo = (const float*)d_in[8];

  const size_t XB_BYTES = (size_t)MROWS * HIDDEN * sizeof(short);   // 8 MiB
  const size_t W_BYTES  = (size_t)HIDDEN * HIDDEN * sizeof(short);  // 2 MiB

  char* ws = (char*)d_ws;
  __bf16* xb  = (__bf16*)ws; ws += XB_BYTES;
  __bf16* wqT = (__bf16*)ws; ws += W_BYTES;
  __bf16* wkT = (__bf16*)ws; ws += W_BYTES;
  __bf16* wvT = (__bf16*)ws; ws += W_BYTES;
  __bf16* woT = (__bf16*)ws; ws += W_BYTES;
  __bf16* qd  = (__bf16*)ws; ws += XB_BYTES;  // [B,H,S,Hd] (pre-scaled)
  __bf16* kd  = (__bf16*)ws; ws += XB_BYTES;  // [B,H,S,Hd]
  __bf16* vtd = (__bf16*)ws; ws += XB_BYTES;  // [B,H,Hd,S]
  __bf16* ctx = (__bf16*)ws; ws += XB_BYTES;  // [B,S,H*Hd]

  // Pre-pass: bf16 conversions (x) and transposed bf16 weights.
  const int nx = MROWS * HIDDEN;
  convert_x_kernel<<<(nx + 255) / 256, 256, 0, stream>>>(x, xb, nx);
  const int nw = HIDDEN * HIDDEN;
  transpose_w_kernel<<<(nw + 255) / 256, 256, 0, stream>>>(wq, wqT);
  transpose_w_kernel<<<(nw + 255) / 256, 256, 0, stream>>>(wk, wkT);
  transpose_w_kernel<<<(nw + 255) / 256, 256, 0, stream>>>(wv, wvT);
  transpose_w_kernel<<<(nw + 255) / 256, 256, 0, stream>>>(wo, woT);

  // Projections (Q gets 1/sqrt(Hd) folded into its epilogue).
  dim3 ggrid(MROWS / 128, HIDDEN / 128);   // 32 x 8
  gemm_bf16_kernel<0><<<ggrid, 256, 0, stream>>>(xb, wqT, bq, qd, ATT_SCALE);
  gemm_bf16_kernel<0><<<ggrid, 256, 0, stream>>>(xb, wkT, bk, kd, 1.0f);
  gemm_bf16_kernel<1><<<ggrid, 256, 0, stream>>>(xb, wvT, bv, vtd, 1.0f);

  // Attention.
  dim3 agrid(SEQ / 128, BATCH * HEADS);    // 16 x 32
  attn_kernel<<<agrid, 256, 0, stream>>>(qd, kd, vtd, ctx);

  // Output projection (fp32 + bias straight to d_out).
  gemm_bf16_kernel<2><<<ggrid, 256, 0, stream>>>(ctx, woT, bo, d_out, 1.0f);
}
